// DivergenceAlignedAttention_75256416960844
// MI455X (gfx1250) — matrix operand order repaired and checked by reading, not compiled, not run on hardware
//
#include <hip/hip_runtime.h>
#include <hip/hip_bf16.h>

// ---------------------------------------------------------------------------
// DivergenceAlignedAttention for MI455X (gfx1250, wave32, WMMA).
//
// Math note: the reference's KL bias adds a per-row constant to scores before
// softmax; softmax is shift-invariant => bias is a no-op on the output.
// So: 4 GEMM projections + QK^T + streaming softmax + PV.
//
// Roofline: ~100 GFLOP vs ~80MB traffic => compute bound => bf16 WMMA
// (v_wmma_f32_16x16x32_bf16, f32 accum). All operands pre-converted to bf16
// so every GEMM inner loop is pure {global_load_b128 -> v_wmma}.
// ---------------------------------------------------------------------------

typedef __bf16 bf16;
typedef __attribute__((ext_vector_type(16))) __bf16 v16bf;
typedef __attribute__((ext_vector_type(8)))  __bf16 v8bf;
typedef __attribute__((ext_vector_type(8)))  float  v8f;
typedef __attribute__((ext_vector_type(4)))  float  v4f;

#define WMMA_BF16(A, B, C) \
  __builtin_amdgcn_wmma_f32_16x16x32_bf16(false, (A), false, (B), (short)0, (C), false, false)

// Problem constants
#define BB 2
#define SS 2048
#define DD 1024
#define HH 16
#define HDD 64

// A-operand (16x32 bf16): lane m = L%16; element e holds
// K = 16*half + e.  Two contiguous v8bf (b128) loads.
__device__ __forceinline__ v16bf load_a16(const bf16* __restrict__ base, int half) {
  v8bf lo = *(const v8bf*)(base + 16 * half);
  v8bf hi = *(const v8bf*)(base + 16 * half + 8);
  v16bf a;
#pragma unroll
  for (int i = 0; i < 8; ++i) { a[i] = lo[i]; a[8 + i] = hi[i]; }
  return a;
}

// B-operand (32x16 bf16): lane n = L%16; element e holds K = 16*half + e.
// One 16-element contiguous run => two v8bf (b128) loads.
__device__ __forceinline__ v16bf load_b16(const bf16* __restrict__ base, int half) {
  v8bf lo = *(const v8bf*)(base + 16 * half);
  v8bf hi = *(const v8bf*)(base + 16 * half + 8);
  v16bf b;
#pragma unroll
  for (int i = 0; i < 8; ++i) { b[i] = lo[i]; b[8 + i] = hi[i]; }
  return b;
}

// ---------------------------------------------------------------------------
// Kernel 0a: fp32 weight W[k][n] -> bf16 transposed WT[n][k].
// ---------------------------------------------------------------------------
__global__ __launch_bounds__(256) void wconv_kernel(const float* __restrict__ W,
                                                    bf16* __restrict__ WT) {
  int id = blockIdx.x * 256 + threadIdx.x;   // 0 .. D*D-1
  int k = id >> 10;
  int n = id & (DD - 1);
  WT[(size_t)n * DD + k] = (bf16)W[(size_t)k * DD + n];
}

// ---------------------------------------------------------------------------
// Kernel 0b: fp32 activations -> bf16 (same layout), 4 elements per thread.
// ---------------------------------------------------------------------------
__global__ __launch_bounds__(256) void xconv_kernel(const float* __restrict__ X,
                                                    bf16* __restrict__ Y) {
  int id = blockIdx.x * 256 + threadIdx.x;   // 0 .. B*S*D/4 - 1
  v4f x = ((const v4f*)X)[id];
#pragma unroll
  for (int i = 0; i < 4; ++i) Y[4 * (size_t)id + i] = (bf16)x[i];
}

// ---------------------------------------------------------------------------
// Kernel 1: GEMM projection  out = Xbf(4096x1024 bf16) @ W + bias.
// One wave computes a 32x64 output tile (2 M-tiles x 4 N-tiles): per K-step
// of 32 that is 8 WMMAs against 12 b128 loads, with pure bf16 operands.
// mode 0: store bf16 at [b,h,s,hd]      (Q, K)
// mode 1: store bf16 at [b,h,hd,s]      (V transposed)
// mode 2: store fp32 at [row, d]        (final output projection)
// ---------------------------------------------------------------------------
__global__ __launch_bounds__(256) void proj_kernel(const bf16* __restrict__ Xbf,
                                                   const bf16* __restrict__ WT,
                                                   const float* __restrict__ bias,
                                                   void* __restrict__ out, int mode) {
  const int wid  = threadIdx.x >> 5;
  const int lane = threadIdx.x & 31;
  const int half = lane >> 4;
  const int n16  = lane & 15;

  const int tile = blockIdx.x * 8 + wid;   // 2048 wave-tiles (128 x 16)
  const int mt2  = tile >> 4;              // 32-row block
  const int nt4  = tile & 15;              // 64-col block

  const bf16* a0row = Xbf + (size_t)(mt2 * 32 + n16) * DD;        // A: lane row m=lane%16
  const bf16* a1row = a0row + (size_t)16 * DD;
  const bf16* wrow0 = WT + (size_t)(nt4 * 64 + n16) * DD;         // B: lane col n=lane%16
  const bf16* wrow1 = wrow0 + (size_t)16 * DD;
  const bf16* wrow2 = wrow0 + (size_t)32 * DD;
  const bf16* wrow3 = wrow0 + (size_t)48 * DD;

  v8f acc[2][4] = {};
  for (int kb = 0; kb < DD; kb += 32) {
    const v16bf A0 = load_a16(a0row + kb, half);
    const v16bf A1 = load_a16(a1row + kb, half);
    const v16bf B0 = load_b16(wrow0 + kb, half);
    const v16bf B1 = load_b16(wrow1 + kb, half);
    const v16bf B2 = load_b16(wrow2 + kb, half);
    const v16bf B3 = load_b16(wrow3 + kb, half);
    acc[0][0] = WMMA_BF16(A0, B0, acc[0][0]);
    acc[0][1] = WMMA_BF16(A0, B1, acc[0][1]);
    acc[0][2] = WMMA_BF16(A0, B2, acc[0][2]);
    acc[0][3] = WMMA_BF16(A0, B3, acc[0][3]);
    acc[1][0] = WMMA_BF16(A1, B0, acc[1][0]);
    acc[1][1] = WMMA_BF16(A1, B1, acc[1][1]);
    acc[1][2] = WMMA_BF16(A1, B2, acc[1][2]);
    acc[1][3] = WMMA_BF16(A1, B3, acc[1][3]);
  }

#pragma unroll
  for (int j = 0; j < 4; ++j) {
    const int d  = nt4 * 64 + j * 16 + n16;   // C layout: lane's column fixed
    const float bv = bias[d];
    const int h  = d >> 6;
    const int hd = d & 63;
#pragma unroll
    for (int i = 0; i < 2; ++i) {
#pragma unroll
      for (int r = 0; r < 8; ++r) {
        const int grow = mt2 * 32 + i * 16 + r + 8 * half;   // C layout: m = r+8*half
        const int b = grow >> 11;                            // row = b*S + s
        const int s = grow & (SS - 1);
        const float val = acc[i][j][r] + bv;
        if (mode == 0) {
          ((bf16*)out)[(((size_t)(b * HH + h)) * SS + s) * HDD + hd] = (bf16)val;
        } else if (mode == 1) {
          ((bf16*)out)[(((size_t)(b * HH + h)) * HDD + hd) * SS + s] = (bf16)val;
        } else {
          ((float*)out)[(size_t)grow * DD + d] = val;
        }
      }
    }
  }
}

// ---------------------------------------------------------------------------
// Kernel 2: flash attention. One wave handles 16 queries of one (b,h).
// Computes S^T = K_tile (A) x Q^T (B) so score tiles land in the layout
// needed for O^T = V^T (A) x P^T (B) with a single half-swap shfl.
// Writes ctx as bf16 row-major so the final projection needs no conversion.
// ---------------------------------------------------------------------------
__global__ __launch_bounds__(256) void attn_kernel(const bf16* __restrict__ Qb,
                                                   const bf16* __restrict__ Kb,
                                                   const bf16* __restrict__ Vt,
                                                   bf16* __restrict__ ctx) {
  const int wid  = threadIdx.x >> 5;
  const int lane = threadIdx.x & 31;
  const int half = lane >> 4;
  const int n16  = lane & 15;

  const int qt = blockIdx.x * 8 + wid;           // 4096 q-tiles total
  const int b  = qt >> 11;                       // 2048 tiles per batch (H * S/16)
  const int h  = (qt >> 7) & 15;
  const int q0 = (qt & 127) * 16;
  const size_t bh = (size_t)(b * HH + h);

  // Q^T as B-operand: lane n = query q0+n16; K dim = hd (two x32 chunks)
  const bf16* qbase = Qb + (bh * SS + q0 + n16) * HDD;
  const v16bf QB0 = load_b16(qbase, half);        // hd 0..31
  const v16bf QB1 = load_b16(qbase + 32, half);   // hd 32..63

  float mi = -1e30f;
  float li = 0.0f;
  v8f O[4] = {};                                  // O^T tiles: hd chunks of 16
  const float scale = 0.125f;                     // 1/sqrt(64)

  for (int kb = 0; kb < SS; kb += 32) {
    // K as A-operand: lane m = local key (n16); K dim = hd
    const bf16* k0 = Kb + (bh * SS + kb + n16) * HDD;
    const bf16* k1 = Kb + (bh * SS + kb + 16 + n16) * HDD;
    const v16bf KA00 = load_a16(k0, half);        // keys kb..kb+15,  hd 0..31
    const v16bf KA01 = load_a16(k0 + 32, half);   //                  hd 32..63
    const v16bf KA10 = load_a16(k1, half);        // keys kb+16..kb+31
    const v16bf KA11 = load_a16(k1 + 32, half);

    v8f s0 = {}, s1 = {};
    s0 = WMMA_BF16(KA00, QB0, s0);
    s0 = WMMA_BF16(KA01, QB1, s0);
    s1 = WMMA_BF16(KA10, QB0, s1);
    s1 = WMMA_BF16(KA11, QB1, s1);

    // S^T C-layout: lane column = query n16; rows = key offsets {8*half+r} (+16 tile1)
    float mloc = -1e30f;
#pragma unroll
    for (int r = 0; r < 8; ++r) {
      s0[r] *= scale; s1[r] *= scale;
      mloc = fmaxf(mloc, fmaxf(s0[r], s1[r]));
    }
    mloc = fmaxf(mloc, __shfl_xor(mloc, 16, 32));   // combine key halves
    const float mnew  = fmaxf(mi, mloc);
    const float alpha = __expf(mi - mnew);

    float p0[8], p1[8];
    float sum = 0.0f;
#pragma unroll
    for (int r = 0; r < 8; ++r) {
      p0[r] = __expf(s0[r] - mnew);
      p1[r] = __expf(s1[r] - mnew);
      sum += p0[r] + p1[r];
    }
    sum += __shfl_xor(sum, 16, 32);
    li = li * alpha + sum;
    mi = mnew;

    // Pack P^T into B-operand layout: lane n = query; element e holds key 16*half+e.
    v16bf PB;
#pragma unroll
    for (int r = 0; r < 8; ++r) {
      const float own  = half ? p1[r] : p0[r];
      const float send = half ? p0[r] : p1[r];
      const float recv = __shfl_xor(send, 16, 32);
      const float lo = half ? recv : own;   // element r     (keys 16*half + 0..7)
      const float hi = half ? own  : recv;  // element 8+r   (keys 16*half + 8..15)
      PB[r]     = (bf16)lo;
      PB[8 + r] = (bf16)hi;
    }

    // O^T += V^T x P^T ; rescale running accumulators by alpha first.
#pragma unroll
    for (int t = 0; t < 4; ++t) {
      const bf16* vr = Vt + (bh * HDD + 16 * t + n16) * SS + kb;  // lane m = hd local
      const v16bf VA = load_a16(vr, half);                        // keys kb..kb+31
#pragma unroll
      for (int r = 0; r < 8; ++r) O[t][r] *= alpha;
      O[t] = WMMA_BF16(VA, PB, O[t]);
    }
  }

  const float rl = 1.0f / li;
  const int q = q0 + n16;   // O^T C-layout: lane column = query
#pragma unroll
  for (int t = 0; t < 4; ++t) {
#pragma unroll
    for (int r = 0; r < 8; ++r) {
      const int hd = 16 * t + r + 8 * half;
      ctx[((size_t)(b * SS + q)) * DD + h * HDD + hd] = (bf16)(O[t][r] * rl);
    }
  }
}

// ---------------------------------------------------------------------------
// Host launcher
// ---------------------------------------------------------------------------
extern "C" void kernel_launch(void* const* d_in, const int* in_sizes, int n_in,
                              void* d_out, int out_size, void* d_ws, size_t ws_size,
                              hipStream_t stream) {
  (void)in_sizes; (void)n_in; (void)out_size;

  const float* query = (const float*)d_in[0];
  const float* key   = (const float*)d_in[1];
  const float* value = (const float*)d_in[2];
  const float* Wq = (const float*)d_in[3];
  const float* bq = (const float*)d_in[4];
  const float* Wk = (const float*)d_in[5];
  const float* bk = (const float*)d_in[6];
  const float* Wv = (const float*)d_in[7];
  const float* bv = (const float*)d_in[8];
  const float* Wo = (const float*)d_in[9];
  const float* bo = (const float*)d_in[10];
  float* out = (float*)d_out;

  // Workspace layout (bytes)
  const size_t WBYTES = (size_t)DD * DD * sizeof(bf16);        // 2 MB per weight
  const size_t TBYTES = (size_t)BB * SS * DD * sizeof(bf16);   // 8 MB per tensor
  if (ws_size < 4 * WBYTES + 7 * TBYTES) return;               // 64 MB total

  char* ws = (char*)d_ws;
  bf16* wtq = (bf16*)(ws + 0 * WBYTES);
  bf16* wtk = (bf16*)(ws + 1 * WBYTES);
  bf16* wtv = (bf16*)(ws + 2 * WBYTES);
  bf16* wto = (bf16*)(ws + 3 * WBYTES);
  char* tb  = ws + 4 * WBYTES;
  bf16* x_q  = (bf16*)(tb + 0 * TBYTES);   // query in bf16
  bf16* x_k  = (bf16*)(tb + 1 * TBYTES);   // key   in bf16
  bf16* x_v  = (bf16*)(tb + 2 * TBYTES);   // value in bf16
  bf16* q_bf = (bf16*)(tb + 3 * TBYTES);   // Q  [b,h,s,hd]
  bf16* k_bf = (bf16*)(tb + 4 * TBYTES);   // K  [b,h,s,hd]
  bf16* v_t  = (bf16*)(tb + 5 * TBYTES);   // V^T [b,h,hd,s]
  bf16* ctxb = (bf16*)(tb + 6 * TBYTES);   // attention output, row-major bf16

  // 0) convert weights (transpose) and activations to bf16
  const int wcBlocks = (DD * DD) / 256;            // 4096
  wconv_kernel<<<wcBlocks, 256, 0, stream>>>(Wq, wtq);
  wconv_kernel<<<wcBlocks, 256, 0, stream>>>(Wk, wtk);
  wconv_kernel<<<wcBlocks, 256, 0, stream>>>(Wv, wtv);
  wconv_kernel<<<wcBlocks, 256, 0, stream>>>(Wo, wto);
  const int xcBlocks = (BB * SS * DD) / (4 * 256); // 4096
  xconv_kernel<<<xcBlocks, 256, 0, stream>>>(query, x_q);
  xconv_kernel<<<xcBlocks, 256, 0, stream>>>(key,   x_k);
  xconv_kernel<<<xcBlocks, 256, 0, stream>>>(value, x_v);

  // 1) projections: 2048 wave-tiles of 32x64, 8 waves per 256-thread block
  const int projBlocks = ((BB * SS) / 32) * (DD / 64) / 8;   // 256
  proj_kernel<<<projBlocks, 256, 0, stream>>>(x_q, wtq, bq, (void*)q_bf, 0);
  proj_kernel<<<projBlocks, 256, 0, stream>>>(x_k, wtk, bk, (void*)k_bf, 0);
  proj_kernel<<<projBlocks, 256, 0, stream>>>(x_v, wtv, bv, (void*)v_t,  1);

  // 2) attention: 4096 wave-tiles (16 queries each), 8 per block
  const int attnBlocks = (BB * HH * (SS / 16)) / 8;          // 512
  attn_kernel<<<attnBlocks, 256, 0, stream>>>(q_bf, k_bf, v_t, ctxb);

  // 3) output projection straight into d_out (fp32)
  proj_kernel<<<projBlocks, 256, 0, stream>>>(ctxb, wto, bo, (void*)out, 2);
}
